// EPMoE_14285061226497
// MI455X (gfx1250) — 8-bit matrix operand order repaired and checked by reading, compiled, not run on hardware
//
#include <hip/hip_runtime.h>
#include <cstdint>


typedef int   v16i __attribute__((ext_vector_type(16)));
typedef float v8f  __attribute__((ext_vector_type(8)));
typedef int   v4i_vs __attribute__((vector_size(16)));

static constexpr int   T_TOK  = 1024;
static constexpr int   H_DIM  = 1024;
static constexpr int   I_DIM  = 2816;
static constexpr int   E_NUM  = 8;
static constexpr float FP8_MAX = 448.0f;

#if __has_builtin(__builtin_amdgcn_global_load_async_to_lds_b128) && \
    __has_builtin(__builtin_amdgcn_s_wait_asynccnt)
#define USE_ASYNC_LDS 1
#else
#define USE_ASYNC_LDS 0
#endif

typedef __attribute__((address_space(1))) v4i_vs gv4i_t;
typedef __attribute__((address_space(3))) v4i_vs lv4i_t;

__device__ __forceinline__ void copy16_to_lds(const unsigned char* gsrc,
                                              unsigned char* lds_dst) {
#if USE_ASYNC_LDS
  __builtin_amdgcn_global_load_async_to_lds_b128(
      (gv4i_t*)(unsigned long long)gsrc,
      (lv4i_t*)(unsigned int)(unsigned long long)lds_dst,
      0, 0);
#else
  *(uint4*)lds_dst = *(const uint4*)gsrc;
#endif
}

template <int N>
__device__ __forceinline__ void async_wait_le() {
#if USE_ASYNC_LDS
  __builtin_amdgcn_s_wait_asynccnt(N);
#endif
}

__device__ __forceinline__ unsigned char f32_to_e4m3(float f) {
  unsigned u = __float_as_uint(f);
  unsigned char s = (unsigned char)((u >> 24) & 0x80u);
  unsigned a = u & 0x7FFFFFFFu;
  if (a >= 0x7F800000u) return (unsigned char)(s | 0x7Fu);
  float af = __uint_as_float(a);
  if (af >= 448.0f) return (unsigned char)(s | 0x7Eu);
  int ei = (int)(a >> 23) - 127;
  if (ei < -6) {
    int q = (int)rintf(af * 512.0f);
    if (q >= 8) return (unsigned char)(s | 0x08u);
    return (unsigned char)(s | (unsigned)q);
  }
  unsigned mant = a & 0x7FFFFFu;
  unsigned m3  = mant >> 20;
  unsigned rem = mant & 0xFFFFFu;
  if (rem > 0x80000u || (rem == 0x80000u && (m3 & 1u))) ++m3;
  if (m3 == 8u) { m3 = 0u; ++ei; }
  if (ei > 8) return (unsigned char)(s | 0x7Eu);
  return (unsigned char)(s | ((unsigned)(ei + 7) << 3) | m3);
}

__device__ __forceinline__ float silu_f(float x) { return x / (1.0f + __expf(-x)); }

__device__ __forceinline__ v16i load_a_frag(const unsigned char* Atile, int lm, int lhalf) {
  v16i a;
#pragma unroll
  for (int c = 0; c < 2; ++c) {
#pragma unroll
    for (int g = 0; g < 4; ++g) {
      const int* p = (const int*)(Atile + lm * 128 + c * 64 + (g >> 1) * 32 + lhalf * 16 + (g & 1) * 8);
      a[c * 8 + g * 2]     = p[0];
      a[c * 8 + g * 2 + 1] = p[1];
    }
  }
  return a;
}

__device__ __forceinline__ v16i load_b_frag(const unsigned char* Brow, int lhalf) {
  v16i b;
#pragma unroll
  for (int g = 0; g < 4; ++g) {
    const int4 v = *(const int4*)(Brow + g * 32 + lhalf * 16);
    b[4 * g]     = v.x;
    b[4 * g + 1] = v.y;
    b[4 * g + 2] = v.z;
    b[4 * g + 3] = v.w;
  }
  return b;
}

__global__ void moe_init(float* __restrict__ out, int nout,
                         int* __restrict__ counts,
                         unsigned* __restrict__ amax13,
                         unsigned* __restrict__ amax2,
                         float* __restrict__ row_w, int nrow) {
  for (int i = blockIdx.x * blockDim.x + threadIdx.x; i < nout;
       i += gridDim.x * blockDim.x) {
    out[i] = 0.0f;
    if (i < nrow) row_w[i] = 0.0f;
    if (i < E_NUM) { counts[i] = 0; amax13[i] = 0u; amax2[i] = 0u; }
  }
}

__global__ void moe_route(const float* __restrict__ logits,
                          int* __restrict__ counts,
                          int* __restrict__ row_token,
                          float* __restrict__ row_w) {
  const int t = blockIdx.x * blockDim.x + threadIdx.x;
  if (t >= T_TOK) return;
  float p[E_NUM];
  float mx = -3.4e38f;
#pragma unroll
  for (int e = 0; e < E_NUM; ++e) { p[e] = logits[t * E_NUM + e]; mx = fmaxf(mx, p[e]); }
  float sum = 0.0f;
#pragma unroll
  for (int e = 0; e < E_NUM; ++e) { p[e] = __expf(p[e] - mx); sum += p[e]; }
  const float inv = 1.0f / sum;
#pragma unroll
  for (int e = 0; e < E_NUM; ++e) p[e] *= inv;
  int i1 = 0; float v1 = p[0];
#pragma unroll
  for (int e = 1; e < E_NUM; ++e) if (p[e] > v1) { v1 = p[e]; i1 = e; }
  int i2 = (i1 == 0) ? 1 : 0; float v2 = p[i2];
#pragma unroll
  for (int e = 0; e < E_NUM; ++e) if (e != i1 && p[e] > v2) { v2 = p[e]; i2 = e; }
  const float rs = 1.0f / (v1 + v2);
  int pos1 = atomicAdd(counts + i1, 1);
  row_token[i1 * T_TOK + pos1] = t;  row_w[i1 * T_TOK + pos1] = v1 * rs;
  int pos2 = atomicAdd(counts + i2, 1);
  row_token[i2 * T_TOK + pos2] = t;  row_w[i2 * T_TOK + pos2] = v2 * rs;
}

__global__ void moe_absmax(const float* __restrict__ w, size_t perE,
                           unsigned* __restrict__ amax) {
  const int e = blockIdx.z;
  const float* base = w + (size_t)e * perE;
  float m = 0.0f;
  for (size_t i = (size_t)blockIdx.x * blockDim.x + threadIdx.x; i < perE;
       i += (size_t)gridDim.x * blockDim.x)
    m = fmaxf(m, fabsf(base[i]));
  __shared__ float red[256];
  red[threadIdx.x] = m;
  __syncthreads();
  for (int s = 128; s > 0; s >>= 1) {
    if ((int)threadIdx.x < s)
      red[threadIdx.x] = fmaxf(red[threadIdx.x], red[threadIdx.x + s]);
    __syncthreads();
  }
  if (threadIdx.x == 0) atomicMax(amax + e, __float_as_uint(red[0]));
}

__global__ void moe_quant(const float* __restrict__ w,
                          const unsigned* __restrict__ amax,
                          unsigned char* __restrict__ q,
                          size_t perE, size_t total4) {
  for (size_t i = (size_t)blockIdx.x * blockDim.x + threadIdx.x; i < total4;
       i += (size_t)gridDim.x * blockDim.x) {
    const size_t idx = i * 4;
    const int e = (int)(idx / perE);
    const float inv = FP8_MAX / fmaxf(__uint_as_float(amax[e]), 1e-20f);
    const float4 v = *(const float4*)(w + idx);
    unsigned packed = (unsigned)f32_to_e4m3(v.x * inv)
                    | ((unsigned)f32_to_e4m3(v.y * inv) << 8)
                    | ((unsigned)f32_to_e4m3(v.z * inv) << 16)
                    | ((unsigned)f32_to_e4m3(v.w * inv) << 24);
    *(unsigned*)(q + idx) = packed;
  }
}

__global__ void moe_gather_x(const float* __restrict__ x,
                             const int* __restrict__ counts,
                             const int* __restrict__ row_token,
                             const float* __restrict__ s13in,
                             unsigned char* __restrict__ xq) {
  const int er = blockIdx.y;
  const int e = er / T_TOK;
  const int r = er % T_TOK;
  if (r >= counts[e]) return;
  const int tok = row_token[er];
  const float inv = 1.0f / s13in[e];
  const int h = (int)threadIdx.x * 4;
  const float4 v = *(const float4*)(x + (size_t)tok * H_DIM + h);
  unsigned packed = (unsigned)f32_to_e4m3(v.x * inv)
                  | ((unsigned)f32_to_e4m3(v.y * inv) << 8)
                  | ((unsigned)f32_to_e4m3(v.z * inv) << 16)
                  | ((unsigned)f32_to_e4m3(v.w * inv) << 24);
  *(unsigned*)(xq + (size_t)er * H_DIM + h) = packed;
}

__global__ void __launch_bounds__(128) moe_gemm1(
    const unsigned char* __restrict__ xq,
    const unsigned char* __restrict__ w13q,
    const unsigned* __restrict__ amax13,
    const float* __restrict__ s13in,
    const float* __restrict__ s2in,
    const int* __restrict__ counts,
    unsigned char* __restrict__ actq) {
  __shared__ unsigned char Atile[2][16 * 128];
  const int e    = blockIdx.z;
  const int cnt  = counts[e];
  const int row0 = blockIdx.y * 16;
  if (row0 >= cnt) return;
  const int wave  = threadIdx.x >> 5;
  const int lane  = threadIdx.x & 31;
  const int lhalf = lane >> 4;
  const int lm    = lane & 15;
  const int n0    = (blockIdx.x * 4 + wave) * 16;

  const unsigned char* Abase = xq + ((size_t)(e * T_TOK + row0)) * H_DIM;
  const unsigned char* Bg = w13q + ((size_t)e * (2 * I_DIM) + (n0 + lm)) * H_DIM;
  const unsigned char* Bu = Bg + (size_t)I_DIM * H_DIM;

  const int am = threadIdx.x >> 3;
  const int ac = threadIdx.x & 7;
  const unsigned char* Asrc = Abase + (size_t)am * H_DIM + ac * 16;
  unsigned char* Adst[2] = { &Atile[0][am * 128 + ac * 16],
                             &Atile[1][am * 128 + ac * 16] };

  v8f cg, cu;
#pragma unroll
  for (int i = 0; i < 8; ++i) { cg[i] = 0.0f; cu[i] = 0.0f; }

  constexpr int KT = H_DIM / 128;
  copy16_to_lds(Asrc, Adst[0]);
  for (int kt = 0; kt < KT; ++kt) {
    const int k0 = kt * 128;
    if (kt + 1 < KT) {
      copy16_to_lds(Asrc + k0 + 128, Adst[(kt + 1) & 1]);
      async_wait_le<1>();
    } else {
      async_wait_le<0>();
    }
    __syncthreads();
    __builtin_prefetch(Bg + k0 + 128, 0, 3);
    __builtin_prefetch(Bu + k0 + 128, 0, 3);
    const v16i a  = load_a_frag(Atile[kt & 1], lm, lhalf);
    const v16i bg = load_b_frag(Bg + k0, lhalf);
    const v16i bu = load_b_frag(Bu + k0, lhalf);
    cg = __builtin_amdgcn_wmma_f32_16x16x128_fp8_fp8(a, bg, (short)0, cg, false, false);
    cu = __builtin_amdgcn_wmma_f32_16x16x128_fp8_fp8(a, bu, (short)0, cu, false, false);
    __syncthreads();
  }

  const float s13   = (__uint_as_float(amax13[e]) * (1.0f / FP8_MAX)) * s13in[e];
  const float invs2 = 1.0f / s2in[e];
  unsigned char* arow = actq + (size_t)(e * T_TOK) * I_DIM;
#pragma unroll
  for (int r = 0; r < 8; ++r) {
    const int row = row0 + r + 8 * lhalf;
    const float g  = cg[r] * s13;
    const float uu = cu[r] * s13;
    arow[(size_t)row * I_DIM + (n0 + lm)] = f32_to_e4m3(silu_f(g) * uu * invs2);
  }
}

__global__ void __launch_bounds__(128) moe_gemm2(
    const unsigned char* __restrict__ actq,
    const unsigned char* __restrict__ w2q,
    const unsigned* __restrict__ amax2,
    const float* __restrict__ s2in,
    const int* __restrict__ counts,
    const int* __restrict__ row_token,
    const float* __restrict__ row_w,
    float* __restrict__ out) {
  __shared__ unsigned char Atile[2][16 * 128];
  const int e    = blockIdx.z;
  const int cnt  = counts[e];
  const int row0 = blockIdx.y * 16;
  if (row0 >= cnt) return;
  const int wave  = threadIdx.x >> 5;
  const int lane  = threadIdx.x & 31;
  const int lhalf = lane >> 4;
  const int lm    = lane & 15;
  const int n0    = (blockIdx.x * 4 + wave) * 16;

  const unsigned char* Abase = actq + ((size_t)(e * T_TOK + row0)) * I_DIM;
  const unsigned char* Brow  = w2q + ((size_t)e * H_DIM + (n0 + lm)) * I_DIM;

  const int am = threadIdx.x >> 3;
  const int ac = threadIdx.x & 7;
  const unsigned char* Asrc = Abase + (size_t)am * I_DIM + ac * 16;
  unsigned char* Adst[2] = { &Atile[0][am * 128 + ac * 16],
                             &Atile[1][am * 128 + ac * 16] };

  v8f c;
#pragma unroll
  for (int i = 0; i < 8; ++i) c[i] = 0.0f;

  constexpr int KT = I_DIM / 128;
  copy16_to_lds(Asrc, Adst[0]);
  for (int kt = 0; kt < KT; ++kt) {
    const int k0 = kt * 128;
    if (kt + 1 < KT) {
      copy16_to_lds(Asrc + k0 + 128, Adst[(kt + 1) & 1]);
      async_wait_le<1>();
    } else {
      async_wait_le<0>();
    }
    __syncthreads();
    __builtin_prefetch(Brow + k0 + 128, 0, 3);
    const v16i a = load_a_frag(Atile[kt & 1], lm, lhalf);
    const v16i b = load_b_frag(Brow + k0, lhalf);
    c = __builtin_amdgcn_wmma_f32_16x16x128_fp8_fp8(a, b, (short)0, c, false, false);
    __syncthreads();
  }

  const float s2 = (__uint_as_float(amax2[e]) * (1.0f / FP8_MAX)) * s2in[e];
#pragma unroll
  for (int r = 0; r < 8; ++r) {
    const int row = row0 + r + 8 * lhalf;
    if (row < cnt) {
      const int tok   = row_token[e * T_TOK + row];
      const float wgt = row_w[e * T_TOK + row];
      atomicAdd(out + (size_t)tok * H_DIM + (n0 + lm), c[r] * s2 * wgt);
    }
  }
}

extern "C" void kernel_launch(void* const* d_in, const int* in_sizes, int n_in,
                              void* d_out, int out_size, void* d_ws, size_t ws_size,
                              hipStream_t stream) {
  (void)in_sizes; (void)n_in; (void)out_size; (void)ws_size;
  const float* x      = (const float*)d_in[0];
  const float* logits = (const float*)d_in[1];
  const float* w13    = (const float*)d_in[2];
  const float* w2     = (const float*)d_in[3];
  const float* s13in  = (const float*)d_in[4];
  const float* s2in   = (const float*)d_in[5];
  float* out = (float*)d_out;

  char* ws = (char*)d_ws;
  const size_t OFF_COUNTS = 0;
  const size_t OFF_AMAX13 = 256;
  const size_t OFF_AMAX2  = 512;
  const size_t OFF_RTOK   = 768;
  const size_t OFF_RW     = OFF_RTOK + (size_t)E_NUM * T_TOK * 4;
  const size_t OFF_XQ     = (OFF_RW + (size_t)E_NUM * T_TOK * 4 + 255) & ~(size_t)255;
  const size_t OFF_W13Q   = OFF_XQ + (size_t)E_NUM * T_TOK * H_DIM;
  const size_t OFF_W2Q    = OFF_W13Q + (size_t)E_NUM * 2 * I_DIM * H_DIM;
  const size_t OFF_ACTQ   = OFF_W2Q + (size_t)E_NUM * H_DIM * I_DIM;

  int*           counts    = (int*)(ws + OFF_COUNTS);
  unsigned*      amax13    = (unsigned*)(ws + OFF_AMAX13);
  unsigned*      amax2     = (unsigned*)(ws + OFF_AMAX2);
  int*           row_token = (int*)(ws + OFF_RTOK);
  float*         row_w     = (float*)(ws + OFF_RW);
  unsigned char* xq        = (unsigned char*)(ws + OFF_XQ);
  unsigned char* w13q      = (unsigned char*)(ws + OFF_W13Q);
  unsigned char* w2q       = (unsigned char*)(ws + OFF_W2Q);
  unsigned char* actq      = (unsigned char*)(ws + OFF_ACTQ);

  const size_t perE13 = (size_t)2 * I_DIM * H_DIM;
  const size_t perE2  = (size_t)H_DIM * I_DIM;

  moe_init<<<dim3(1024), 256, 0, stream>>>(out, T_TOK * H_DIM, counts, amax13,
                                           amax2, row_w, E_NUM * T_TOK);
  moe_route<<<dim3(4), 256, 0, stream>>>(logits, counts, row_token, row_w);

  moe_absmax<<<dim3(256, 1, E_NUM), 256, 0, stream>>>(w13, perE13, amax13);
  moe_absmax<<<dim3(256, 1, E_NUM), 256, 0, stream>>>(w2,  perE2,  amax2);

  moe_quant<<<dim3(4096), 256, 0, stream>>>(w13, amax13, w13q, perE13,
                                            (size_t)E_NUM * perE13 / 4);
  moe_quant<<<dim3(4096), 256, 0, stream>>>(w2, amax2, w2q, perE2,
                                            (size_t)E_NUM * perE2 / 4);

  moe_gather_x<<<dim3(1, E_NUM * T_TOK), 256, 0, stream>>>(x, counts, row_token,
                                                           s13in, xq);

  moe_gemm1<<<dim3(I_DIM / 64, T_TOK / 16, E_NUM), 128, 0, stream>>>(
      xq, w13q, amax13, s13in, s2in, counts, actq);

  moe_gemm2<<<dim3(H_DIM / 64, T_TOK / 16, E_NUM), 128, 0, stream>>>(
      actq, w2q, amax2, s2in, counts, row_token, row_w, out);
}
